// LSTMForwardEncoder_8246337208480
// MI455X (gfx1250) — matrix operand order repaired and checked by reading, not compiled, not run on hardware
//
#include <hip/hip_runtime.h>
#include <hip/hip_bf16.h>

typedef __bf16 bf16;
typedef __attribute__((ext_vector_type(16))) __bf16 v16bf;
typedef __attribute__((ext_vector_type(8)))  __bf16 v8bf;
typedef __attribute__((ext_vector_type(8)))  float  v8f;

#define T_SEQ   512
#define BATCH   128
#define IN_DIM  512
#define H_DIM   512
#define K_DIM   (IN_DIM + H_DIM)      // 1024 (x | h concatenated)
#define G4H     (4 * H_DIM)           // 2048 gate rows
#define LDS_ROW (K_DIM + 8)           // padded LDS row stride (bf16 elems)
#define NBLOCKS (H_DIM / 16)          // 32 persistent workgroups
#define NTHREADS 128                  // 4 waves (wave32)

// workspace layout (bytes)
#define OFF_CNT  0
#define OFF_W    256
#define OFF_BIAS (OFF_W + (size_t)G4H * K_DIM * 2)          // bf16 packed weights (4 MB)
#define OFF_C    (OFF_BIAS + (size_t)G4H * 4)               // combined bias (f32)
#define OFF_HBF  (OFF_C + (size_t)BATCH * H_DIM * 4)        // bf16 h, double-buffered
#define OFF_XBF  (OFF_HBF + (size_t)2 * BATCH * H_DIM * 2)  // bf16 sequence (67 MB)

__device__ __forceinline__ float sigm(float x) { return 1.0f / (1.0f + __expf(-x)); }
__device__ __forceinline__ float tanh_f(float x) {
    float e = __expf(2.0f * x);
    return (e - 1.0f) / (e + 1.0f);
}

// 16x32 bf16 A-fragment slice for this lane, loaded directly as bf16.
// Lane layout (ISA 7.12.2): lanes 0-15 hold K=[0..7] (v0-3) and [16..23] (v4-7),
// lanes 16-31 hold K=[8..15] and [24..31]; caller pre-applies the +0/+8 shift.
__device__ __forceinline__ v16bf load_a_bf(const bf16* __restrict__ S) {
    v8bf lo = *(const v8bf*)S;          // K base .. base+7   (16 B)
    v8bf hi = *(const v8bf*)(S + 16);   // K base+16 .. +23   (16 B)
    return __builtin_shufflevector(lo, hi,
        0,1,2,3,4,5,6,7,8,9,10,11,12,13,14,15);
}

// One-time prep: pack W_ih|W_hh -> bf16 Wc[2048][1024], sum biases,
// convert the whole sequence to bf16, broadcast h0 (bf16) / c0 (f32).
__global__ void lstm_prep(const float* __restrict__ seq,
                          const float* __restrict__ Wih, const float* __restrict__ Whh,
                          const float* __restrict__ bih, const float* __restrict__ bhh,
                          const float* __restrict__ h0,  const float* __restrict__ c0,
                          bf16* __restrict__ Wc, float* __restrict__ bias,
                          float* __restrict__ cbuf, bf16* __restrict__ hbf,
                          bf16* __restrict__ xbf) {
    const int tid  = blockIdx.x * blockDim.x + threadIdx.x;
    const int nthr = gridDim.x * blockDim.x;
    for (size_t idx = tid; idx < (size_t)T_SEQ * BATCH * IN_DIM; idx += nthr)
        xbf[idx] = (bf16)seq[idx];
    for (int idx = tid; idx < G4H * K_DIM; idx += nthr) {
        const int row = idx >> 10;
        const int k   = idx & (K_DIM - 1);
        const float v = (k < IN_DIM) ? Wih[row * IN_DIM + k]
                                     : Whh[row * H_DIM + (k - IN_DIM)];
        Wc[idx] = (bf16)v;
    }
    for (int idx = tid; idx < G4H; idx += nthr) bias[idx] = bih[idx] + bhh[idx];
    for (int idx = tid; idx < BATCH * H_DIM; idx += nthr) {
        cbuf[idx] = c0[idx & (H_DIM - 1)];
        hbf[idx]  = (bf16)h0[idx & (H_DIM - 1)];   // read buffer for t=0
    }
}

// Persistent LSTM: 32 blocks, each owns a 16-wide H slice (all 4 gates) with its
// weight slice resident in LDS; grid-wide atomic barrier between timesteps.
__global__ void __launch_bounds__(NTHREADS)
lstm_persistent(const bf16* __restrict__ xbf, const bf16* __restrict__ Wc,
                const float* __restrict__ bias, float* __restrict__ cbuf,
                bf16* hbf, float* __restrict__ out, unsigned* counter) {
    extern __shared__ char smem_raw[];
    bf16* wlds = (bf16*)smem_raw;        // 64 rows x LDS_ROW bf16 = ~129 KB

    const int lane  = threadIdx.x & 31;
    const int wave  = threadIdx.x >> 5;
    const int nbase = blockIdx.x * 16;   // H-slice base

    // Stage this block's 64 weight rows (gate g, col n -> LDS row g*16+n) once.
    for (int i = threadIdx.x; i < 64 * (K_DIM / 8); i += NTHREADS) {
        const int lrow  = i >> 7;        // 128 16-byte chunks per row
        const int chunk = i & 127;
        const int gate  = lrow >> 4;
        const int n     = lrow & 15;
        const int grow  = gate * H_DIM + nbase + n;
        const uint4* src = (const uint4*)(Wc + (size_t)grow * K_DIM + chunk * 8);
        *(uint4*)(wlds + lrow * LDS_ROW + chunk * 8) = *src;
    }
    __syncthreads();

    const int colLocal = lane & 15;
    const int col      = nbase + colLocal;        // h index this lane produces
    const int hiHalf   = (lane & 16) ? 8 : 0;     // C-tile M offset for upper lanes
    const int aoff     = hiHalf;                  // A-frag K shift for upper lanes
    const int mbase    = wave * 32;               // 2 M-tiles per wave

    float biasv[4];
#pragma unroll
    for (int g = 0; g < 4; ++g) biasv[g] = bias[g * H_DIM + col];

    for (int t = 0; t < T_SEQ; ++t) {
        const bf16* xt     = xbf + (size_t)t * BATCH * IN_DIM;
        const bf16* hread  = hbf + (size_t)(t & 1) * (BATCH * H_DIM);
        bf16*       hwrite = hbf + (size_t)((t + 1) & 1) * (BATCH * H_DIM);

        if (t + 1 < T_SEQ) {  // warm L2 with next timestep's bf16 activations
            const int gid = blockIdx.x * NTHREADS + threadIdx.x;
            __builtin_prefetch(xbf + (size_t)(t + 1) * BATCH * IN_DIM + gid * 16, 0, 1);
        }

        v8f acc[4][2];
        const v8f zero = {0.f, 0.f, 0.f, 0.f, 0.f, 0.f, 0.f, 0.f};
#pragma unroll
        for (int g = 0; g < 4; ++g) { acc[g][0] = zero; acc[g][1] = zero; }

        // gates[128,2048] = [x_t | h_{t-1}] @ Wc^T   (K = 1024, bf16 WMMA, f32 acc)
#pragma unroll 2
        for (int kk = 0; kk < K_DIM; kk += 32) {
            const bf16* sb = (kk < IN_DIM) ? (xt + kk) : (hread + (kk - IN_DIM));
            const bf16* Sa = sb + (size_t)(mbase + colLocal) * 512 + aoff;
            v16bf a0 = load_a_bf(Sa);
            v16bf a1 = load_a_bf(Sa + 16 * 512);
            const int bko = kk + ((lane & 16) ? 8 : 0);
#pragma unroll
            for (int g = 0; g < 4; ++g) {
                const bf16* lp = wlds + (g * 16 + colLocal) * LDS_ROW + bko;
                v8bf lo = *(const v8bf*)lp;
                v8bf hi = *(const v8bf*)(lp + 16);
                v16bf bw = __builtin_shufflevector(lo, hi,
                    0,1,2,3,4,5,6,7,8,9,10,11,12,13,14,15);
                acc[g][0] = __builtin_amdgcn_wmma_f32_16x16x32_bf16(
                    false, a0, false, bw, (short)0, acc[g][0], false, false);
                acc[g][1] = __builtin_amdgcn_wmma_f32_16x16x32_bf16(
                    false, a1, false, bw, (short)0, acc[g][1], false, false);
            }
        }

        // Fused gate nonlinearities + cell/hidden update (i,f,g,o share lane slots).
#pragma unroll
        for (int m = 0; m < 2; ++m) {
#pragma unroll
            for (int r = 0; r < 8; ++r) {
                const int row = mbase + m * 16 + hiHalf + r;     // batch index
                const size_t idx = (size_t)row * H_DIM + col;
                const float iv = acc[0][m][r] + biasv[0];
                const float fv = acc[1][m][r] + biasv[1];
                const float gv = acc[2][m][r] + biasv[2];
                const float ov = acc[3][m][r] + biasv[3];
                const float cold = cbuf[idx];
                const float cnew = sigm(fv) * cold + sigm(iv) * tanh_f(gv);
                const float hv   = sigm(ov) * tanh_f(cnew);
                cbuf[idx] = cnew;
                hwrite[idx] = (bf16)hv;                           // bf16 h for next step
                out[(size_t)t * (BATCH * H_DIM) + idx] = hv;      // f32 result
            }
        }

        // Grid-wide step barrier (monotonic counter; all 32 blocks resident).
        __syncthreads();
        if (threadIdx.x == 0) {
            __threadfence();   // make h_t / c_t visible device-wide
            __hip_atomic_fetch_add(counter, 1u, __ATOMIC_RELEASE, __HIP_MEMORY_SCOPE_AGENT);
            const unsigned target = (unsigned)(t + 1) * gridDim.x;
            while (__hip_atomic_load(counter, __ATOMIC_ACQUIRE, __HIP_MEMORY_SCOPE_AGENT) < target)
                __builtin_amdgcn_s_sleep(1);
        }
        __syncthreads();
        __builtin_amdgcn_fence(__ATOMIC_ACQUIRE, "agent");  // invalidate L0 before reading h_t
    }
}

extern "C" void kernel_launch(void* const* d_in, const int* in_sizes, int n_in,
                              void* d_out, int out_size, void* d_ws, size_t ws_size,
                              hipStream_t stream) {
    (void)in_sizes; (void)n_in; (void)out_size; (void)ws_size;
    const float* seq = (const float*)d_in[0];
    const float* Wih = (const float*)d_in[1];
    const float* Whh = (const float*)d_in[2];
    const float* bih = (const float*)d_in[3];
    const float* bhh = (const float*)d_in[4];
    const float* h0  = (const float*)d_in[5];
    const float* c0  = (const float*)d_in[6];
    float* out = (float*)d_out;

    char* ws = (char*)d_ws;
    unsigned* counter = (unsigned*)(ws + OFF_CNT);
    bf16*  Wc   = (bf16*)(ws + OFF_W);
    float* bias = (float*)(ws + OFF_BIAS);
    float* cbuf = (float*)(ws + OFF_C);
    bf16*  hbf  = (bf16*)(ws + OFF_HBF);
    bf16*  xbf  = (bf16*)(ws + OFF_XBF);

    hipMemsetAsync(ws, 0, 256, stream);  // reset barrier counter every call

    lstm_prep<<<dim3(8192), dim3(256), 0, stream>>>(
        seq, Wih, Whh, bih, bhh, h0, c0, Wc, bias, cbuf, hbf, xbf);

    const size_t smem = (size_t)64 * LDS_ROW * sizeof(bf16);  // ~129 KB of 320 KB/WGP
    lstm_persistent<<<dim3(NBLOCKS), dim3(NTHREADS), smem, stream>>>(
        xbf, Wc, bias, cbuf, hbf, out, counter);
}
